// SwinTransformerBlock_65360812311138
// MI455X (gfx1250) — matrix operand order repaired and checked by reading, not compiled, not run on hardware
//
#include <hip/hip_runtime.h>
#include <hip/hip_bf16.h>
#include <math.h>

#define TOK   100352      // 32 * 56 * 56
#define CDIM  192
#define HEADS 6
#define HD    32
#define NWIN  2048        // 32 * 8 * 8
#define MLP   768

typedef __attribute__((ext_vector_type(16))) _Float16 v16h;
typedef __attribute__((ext_vector_type(8)))  _Float16 v8h;
typedef __attribute__((ext_vector_type(8)))  float    v8f;

static __device__ __forceinline__ v16h frag_cat(v8h a0, v8h a1) {
  v16h r;
#pragma unroll
  for (int i = 0; i < 8; ++i) { r[i] = a0[i]; r[i + 8] = a1[i]; }
  return r;
}

// A fragment (16xK slice, K-step 32) of row-major f16 matrix.
// Lane l holds row (l&15); elements 0-7 = K k+8*hh .. ; elements 8-15 = K k+16+8*hh ..
static __device__ __forceinline__ v16h load_a_frag(const _Float16* A, long row, int lda, int k, int hh) {
  const _Float16* p = A + row * (long)lda + k + 8 * hh;
  return frag_cat(*(const v8h*)p, *(const v8h*)(p + 16));
}

// B fragment (Kx16 slice, K-step 32) read from *transposed* (N x K row-major) storage.
// Lane l holds col n=(l&15); elements 0-7 = K k+8*hh .. ; elements 8-15 = K k+16+8*hh ..
static __device__ __forceinline__ v16h load_b_frag(const _Float16* Bt, long nrow, int ldb, int k, int hh) {
  const _Float16* p = Bt + nrow * (long)ldb + k + 8 * hh;
  return frag_cat(*(const v8h*)p, *(const v8h*)(p + 16));
}

static __device__ __forceinline__ v8f wmma32(v16h a, v16h b, v8f c) {
  return __builtin_amdgcn_wmma_f32_16x16x32_f16(false, a, false, b, (short)0, c, false, false);
}

static __device__ __forceinline__ float wave_sum(float v) {
#pragma unroll
  for (int m = 1; m < 32; m <<= 1) v += __shfl_xor(v, m, 32);
  return v;
}

// ---------------------------------------------------------------- weight prep
__global__ void transpose_to_f16_kernel(const float* __restrict__ W, _Float16* __restrict__ WT,
                                        int K, int N) {
  long idx = (long)blockIdx.x * blockDim.x + threadIdx.x;
  if (idx >= (long)K * N) return;
  long n = idx / K, k = idx % K;
  WT[idx] = (_Float16)W[k * (long)N + n];
}

__global__ void fill_zero16(float4* __restrict__ p, long n16) {
  long i = (long)blockIdx.x * blockDim.x + threadIdx.x;
  if (i < n16) p[i] = make_float4(0.f, 0.f, 0.f, 0.f);
}

// ------------------------------------------------- LN1 + shift + window gather
__global__ void ln1_window_kernel(const float* __restrict__ x, const float* __restrict__ w,
                                  const float* __restrict__ b, _Float16* __restrict__ yw) {
  int t    = (int)(((long)blockIdx.x * blockDim.x + threadIdx.x) >> 5);
  int lane = threadIdx.x & 31;
  int win = t / 49, n = t % 49;
  int bb = win >> 6, wrem = win & 63, wh = wrem >> 3, ww = wrem & 7;
  int i = n / 7, j = n % 7;
  int hr = (wh * 7 + i + 3) % 56;   // un-roll of roll(-3)
  int wc = (ww * 7 + j + 3) % 56;
  long src = ((long)bb * 3136 + (long)hr * 56 + wc) * CDIM;
  float v[6]; float s = 0.f, s2 = 0.f;
#pragma unroll
  for (int q = 0; q < 6; ++q) { v[q] = x[src + lane + 32 * q]; s += v[q]; s2 += v[q] * v[q]; }
  s = wave_sum(s); s2 = wave_sum(s2);
  float mu  = s * (1.0f / CDIM);
  float var = s2 * (1.0f / CDIM) - mu * mu;
  float rs  = rsqrtf(var + 1e-5f);
#pragma unroll
  for (int q = 0; q < 6; ++q) {
    int c = lane + 32 * q;
    yw[(long)t * CDIM + c] = (_Float16)((v[q] - mu) * rs * w[c] + b[c]);
  }
}

// ------------------------------------------------------------------- qkv GEMM
// wave = 16 rows x 64 cols of the 100352x576 GEMM; epilogue scatters q/k/vT.
__global__ void qkv_gemm_kernel(const _Float16* __restrict__ yw, const _Float16* __restrict__ qkvT,
                                const float* __restrict__ qkv_b,
                                _Float16* __restrict__ q, _Float16* __restrict__ k,
                                _Float16* __restrict__ vT) {
  int wid  = (int)(((long)blockIdx.x * blockDim.x + threadIdx.x) >> 5);
  int lane = threadIdx.x & 31;
  int rt = wid / 9, cg = wid % 9;
  long rowBase = (long)rt * 16;
  int colBase = cg * 64;
  int hh = lane >> 4, ln = lane & 15;
  v8f acc[4];
#pragma unroll
  for (int tn = 0; tn < 4; ++tn) acc[tn] = (v8f)(0.f);
#pragma unroll
  for (int kk = 0; kk < CDIM; kk += 32) {
    v16h a = load_a_frag(yw, rowBase + ln, CDIM, kk, hh);
#pragma unroll
    for (int tn = 0; tn < 4; ++tn) {
      v16h bf = load_b_frag(qkvT, colBase + tn * 16 + ln, CDIM, kk, hh);
      acc[tn] = wmma32(a, bf, acc[tn]);
    }
  }
  const float QS = 0.17677669529663687f;  // 32^-0.5
#pragma unroll
  for (int tn = 0; tn < 4; ++tn) {
#pragma unroll
    for (int r = 0; r < 8; ++r) {
      int col = colBase + tn * 16 + ln;
      int three = col / 192, head = (col % 192) / 32, d = col % 32;
      long tok = rowBase + r + 8 * hh;
      long wdw = tok / 49; int ntok = (int)(tok % 49);
      float val = acc[tn][r] + qkv_b[col];
      long wh6 = wdw * HEADS + head;
      if (three == 0)      q [(wh6 * 64 + ntok) * 32 + d] = (_Float16)(val * QS);
      else if (three == 1) k [(wh6 * 64 + ntok) * 32 + d] = (_Float16)val;
      else                 vT[(wh6 * 32 + d) * 64 + ntok] = (_Float16)val;
    }
  }
}

// ------------------------------------------------------------------ attention
// one wave per (window, head): QK^T -> bias -> masked softmax -> P (LDS) -> P.V
__global__ __launch_bounds__(128) void attn_kernel(const _Float16* __restrict__ q,
                                                   const _Float16* __restrict__ k,
                                                   const _Float16* __restrict__ vT,
                                                   const float* __restrict__ rel_bias,
                                                   _Float16* __restrict__ attn_out) {
  __shared__ _Float16 pbuf[4][64 * 64];
  int warp = threadIdx.x >> 5, lane = threadIdx.x & 31;
  int wid = blockIdx.x * 4 + warp;
  int wdw = wid / HEADS, head = wid % HEADS;
  long base = (long)(wdw * HEADS + head);
  const _Float16* qb = q  + base * 64 * 32;
  const _Float16* kb = k  + base * 64 * 32;
  const _Float16* vb = vT + base * 32 * 64;
  int hh = lane >> 4, ln = lane & 15;

  v8f s[4][4];
#pragma unroll
  for (int ti = 0; ti < 4; ++ti)
#pragma unroll
    for (int tj = 0; tj < 4; ++tj) s[ti][tj] = (v8f)(0.f);

  v16h kf[4];
#pragma unroll
  for (int tj = 0; tj < 4; ++tj) kf[tj] = load_b_frag(kb, tj * 16 + ln, 32, 0, hh);
#pragma unroll
  for (int ti = 0; ti < 4; ++ti) {
    v16h af = load_a_frag(qb, ti * 16 + ln, 32, 0, hh);
#pragma unroll
    for (int tj = 0; tj < 4; ++tj) s[ti][tj] = wmma32(af, kf[tj], s[ti][tj]);
  }

  // relative-position bias + mask
#pragma unroll
  for (int ti = 0; ti < 4; ++ti)
#pragma unroll
    for (int tj = 0; tj < 4; ++tj)
#pragma unroll
      for (int r = 0; r < 8; ++r) {
        int m = ti * 16 + r + 8 * hh;
        int n = tj * 16 + ln;
        if (m < 49 && n < 49) {
          int i1 = m / 7, j1 = m % 7, i2 = n / 7, j2 = n % 7;
          int idx = (i1 - i2 + 6) * 13 + (j1 - j2 + 6);
          s[ti][tj][r] += rel_bias[idx * HEADS + head];
        } else {
          s[ti][tj][r] = -1e30f;
        }
      }

  // row softmax (row lives in one 16-lane half across 4 tj tiles)
#pragma unroll
  for (int ti = 0; ti < 4; ++ti)
#pragma unroll
    for (int r = 0; r < 8; ++r) {
      float mx = -1e30f;
#pragma unroll
      for (int tj = 0; tj < 4; ++tj) mx = fmaxf(mx, s[ti][tj][r]);
#pragma unroll
      for (int msk = 1; msk < 16; msk <<= 1) mx = fmaxf(mx, __shfl_xor(mx, msk, 32));
      float sum = 0.f;
#pragma unroll
      for (int tj = 0; tj < 4; ++tj) { float e = __expf(s[ti][tj][r] - mx); s[ti][tj][r] = e; sum += e; }
#pragma unroll
      for (int msk = 1; msk < 16; msk <<= 1) sum += __shfl_xor(sum, msk, 32);
      float inv = 1.f / sum;
#pragma unroll
      for (int tj = 0; tj < 4; ++tj) s[ti][tj][r] *= inv;
    }

  // stage P in LDS as f16 (A-source for P.V)
  _Float16* pl = pbuf[warp];
#pragma unroll
  for (int ti = 0; ti < 4; ++ti)
#pragma unroll
    for (int tj = 0; tj < 4; ++tj)
#pragma unroll
      for (int r = 0; r < 8; ++r)
        pl[(ti * 16 + r + 8 * hh) * 64 + tj * 16 + ln] = (_Float16)s[ti][tj][r];
  asm volatile("s_wait_dscnt 0" ::: "memory");

  v8f o[4][2];
#pragma unroll
  for (int ti = 0; ti < 4; ++ti)
#pragma unroll
    for (int tn = 0; tn < 2; ++tn) o[ti][tn] = (v8f)(0.f);
#pragma unroll
  for (int ti = 0; ti < 4; ++ti)
#pragma unroll
    for (int ks = 0; ks < 2; ++ks) {
      v16h af = load_a_frag(pl, ti * 16 + ln, 64, ks * 32, hh);
#pragma unroll
      for (int tn = 0; tn < 2; ++tn) {
        v16h bf = load_b_frag(vb, tn * 16 + ln, 64, ks * 32, hh);
        o[ti][tn] = wmma32(af, bf, o[ti][tn]);
      }
    }
#pragma unroll
  for (int ti = 0; ti < 4; ++ti)
#pragma unroll
    for (int tn = 0; tn < 2; ++tn)
#pragma unroll
      for (int r = 0; r < 8; ++r) {
        int m = ti * 16 + r + 8 * hh;
        if (m < 49) {
          int d = tn * 16 + ln;
          attn_out[((long)wdw * 49 + m) * CDIM + head * 32 + d] = (_Float16)o[ti][tn][r];
        }
      }
}

// ------------------------------------------------ proj GEMM + unshift + resid
__global__ void proj_gemm_kernel(const _Float16* __restrict__ attn_out, const _Float16* __restrict__ projT,
                                 const float* __restrict__ proj_b, const float* __restrict__ x,
                                 float* __restrict__ y) {
  int wid  = (int)(((long)blockIdx.x * blockDim.x + threadIdx.x) >> 5);
  int lane = threadIdx.x & 31;
  int rt = wid / 3, cg = wid % 3;
  long rowBase = (long)rt * 16;
  int colBase = cg * 64;
  int hh = lane >> 4, ln = lane & 15;
  v8f acc[4];
#pragma unroll
  for (int tn = 0; tn < 4; ++tn) acc[tn] = (v8f)(0.f);
#pragma unroll
  for (int kk = 0; kk < CDIM; kk += 32) {
    v16h a = load_a_frag(attn_out, rowBase + ln, CDIM, kk, hh);
#pragma unroll
    for (int tn = 0; tn < 4; ++tn) {
      v16h bf = load_b_frag(projT, colBase + tn * 16 + ln, CDIM, kk, hh);
      acc[tn] = wmma32(a, bf, acc[tn]);
    }
  }
#pragma unroll
  for (int tn = 0; tn < 4; ++tn)
#pragma unroll
    for (int r = 0; r < 8; ++r) {
      long t = rowBase + r + 8 * hh;
      int col = colBase + tn * 16 + ln;
      long wdw = t / 49; int n = (int)(t % 49);
      int bb = (int)(wdw >> 6), wrem = (int)(wdw & 63), wh = wrem >> 3, ww = wrem & 7;
      int i = n / 7, j = n % 7;
      int hr = (wh * 7 + i + 3) % 56;
      int wc = (ww * 7 + j + 3) % 56;
      long orow = ((long)bb * 3136 + (long)hr * 56 + wc);
      y[orow * CDIM + col] = x[orow * CDIM + col] + acc[tn][r] + proj_b[col];
    }
}

// ------------------------------------- fused LN2 + fc1 + GELU + fc2 + residual
__global__ __launch_bounds__(256) void mlp_kernel(float* __restrict__ y,
                                                  const float* __restrict__ n2w, const float* __restrict__ n2b,
                                                  const _Float16* __restrict__ fc1T, const float* __restrict__ fc1_b,
                                                  const _Float16* __restrict__ fc2T, const float* __restrict__ fc2_b) {
  __shared__ _Float16 hbuf[16 * CDIM];
  __shared__ _Float16 abuf[16 * MLP];
  int warp = threadIdx.x >> 5, lane = threadIdx.x & 31;
  long rowBase = (long)blockIdx.x * 16;

  // LN2: each wave normalizes 2 rows into LDS
#pragma unroll
  for (int rr = 0; rr < 2; ++rr) {
    int row = warp * 2 + rr;
    long gr = rowBase + row;
    float v[6]; float s = 0.f, s2 = 0.f;
#pragma unroll
    for (int q = 0; q < 6; ++q) { v[q] = y[gr * CDIM + lane + 32 * q]; s += v[q]; s2 += v[q] * v[q]; }
    s = wave_sum(s); s2 = wave_sum(s2);
    float mu  = s * (1.0f / CDIM);
    float var = s2 * (1.0f / CDIM) - mu * mu;
    float rs  = rsqrtf(var + 1e-5f);
#pragma unroll
    for (int q = 0; q < 6; ++q) {
      int c = lane + 32 * q;
      hbuf[row * CDIM + c] = (_Float16)((v[q] - mu) * rs * n2w[c] + n2b[c]);
    }
  }
  __syncthreads();

  int hh = lane >> 4, ln = lane & 15;
  // fc1: wave owns 96 output cols = 6 tiles
  v8f acc[6];
#pragma unroll
  for (int t = 0; t < 6; ++t) acc[t] = (v8f)(0.f);
#pragma unroll
  for (int kk = 0; kk < CDIM; kk += 32) {
    v16h a = load_a_frag(hbuf, ln, CDIM, kk, hh);
#pragma unroll
    for (int t = 0; t < 6; ++t) {
      v16h bf = load_b_frag(fc1T, warp * 96 + t * 16 + ln, CDIM, kk, hh);
      acc[t] = wmma32(a, bf, acc[t]);
    }
  }
#pragma unroll
  for (int t = 0; t < 6; ++t)
#pragma unroll
    for (int r = 0; r < 8; ++r) {
      int col = warp * 96 + t * 16 + ln;
      int row = r + 8 * hh;
      float val = acc[t][r] + fc1_b[col];
      val = 0.5f * val * (1.0f + erff(val * 0.7071067811865476f));  // exact GELU
      abuf[row * MLP + col] = (_Float16)val;
    }
  __syncthreads();

  // fc2: 12 tiles distributed over 8 waves
  for (int t2 = warp; t2 < 12; t2 += 8) {
    v8f a2 = (v8f)(0.f);
#pragma unroll
    for (int kk = 0; kk < MLP; kk += 32) {
      v16h a = load_a_frag(abuf, ln, MLP, kk, hh);
      v16h bf = load_b_frag(fc2T, t2 * 16 + ln, MLP, kk, hh);
      a2 = wmma32(a, bf, a2);
    }
#pragma unroll
    for (int r = 0; r < 8; ++r) {
      int col = t2 * 16 + ln;
      long gr = rowBase + r + 8 * hh;
      y[gr * CDIM + col] = y[gr * CDIM + col] + a2[r] + fc2_b[col];
    }
  }
}

// ------------------------------------------------------------------- launcher
extern "C" void kernel_launch(void* const* d_in, const int* in_sizes, int n_in,
                              void* d_out, int out_size, void* d_ws, size_t ws_size,
                              hipStream_t stream) {
  const float* x      = (const float*)d_in[0];
  const float* n1w    = (const float*)d_in[1];
  const float* n1b    = (const float*)d_in[2];
  const float* qkv_w  = (const float*)d_in[3];
  const float* qkv_b  = (const float*)d_in[4];
  const float* relb   = (const float*)d_in[5];
  const float* proj_w = (const float*)d_in[6];
  const float* proj_b = (const float*)d_in[7];
  const float* n2w    = (const float*)d_in[8];
  const float* n2b    = (const float*)d_in[9];
  const float* fc1_w  = (const float*)d_in[10];
  const float* fc1_b  = (const float*)d_in[11];
  const float* fc2_w  = (const float*)d_in[12];
  const float* fc2_b  = (const float*)d_in[13];
  float* y = (float*)d_out;   // residual stream lives in d_out

  char* ws = (char*)d_ws;
  size_t off = 0;
  auto take = [&](size_t bytes) -> char* {
    char* p = ws + off;
    off += (bytes + 255) & ~(size_t)255;
    return p;
  };
  _Float16* qkvT  = (_Float16*)take((size_t)576 * 192 * 2);
  _Float16* projT = (_Float16*)take((size_t)192 * 192 * 2);
  _Float16* fc1T  = (_Float16*)take((size_t)192 * 768 * 2);  // stored as 768x192
  _Float16* fc2T  = (_Float16*)take((size_t)768 * 192 * 2);  // stored as 192x768
  _Float16* qbuf  = (_Float16*)take((size_t)NWIN * HEADS * 64 * 32 * 2);
  _Float16* kbuf  = (_Float16*)take((size_t)NWIN * HEADS * 64 * 32 * 2);
  _Float16* vTbuf = (_Float16*)take((size_t)NWIN * HEADS * 32 * 64 * 2);
  _Float16* ywbuf = (_Float16*)take((size_t)TOK * CDIM * 2);  // reused as attn_out

  // 1. weight transposition/conversion
  transpose_to_f16_kernel<<<(192 * 576 + 255) / 256, 256, 0, stream>>>(qkv_w, qkvT, 192, 576);
  transpose_to_f16_kernel<<<(192 * 192 + 255) / 256, 256, 0, stream>>>(proj_w, projT, 192, 192);
  transpose_to_f16_kernel<<<(192 * 768 + 255) / 256, 256, 0, stream>>>(fc1_w, fc1T, 192, 768);
  transpose_to_f16_kernel<<<(768 * 192 + 255) / 256, 256, 0, stream>>>(fc2_w, fc2T, 768, 192);

  // 2. zero padded q/k/vT (contiguous region)
  long n16 = (long)3 * NWIN * HEADS * 64 * 32 * 2 / 16;
  fill_zero16<<<(unsigned)((n16 + 255) / 256), 256, 0, stream>>>((float4*)qbuf, n16);

  // 3. LN1 + shift + window partition   (100352 waves)
  ln1_window_kernel<<<TOK / 8, 256, 0, stream>>>(x, n1w, n1b, ywbuf);

  // 4. qkv GEMM  (6272 row-tiles x 9 col-groups = 56448 waves)
  qkv_gemm_kernel<<<56448 / 8, 256, 0, stream>>>(ywbuf, qkvT, qkv_b, qbuf, kbuf, vTbuf);

  // 5. attention  (12288 (window,head) waves, 4 per block)
  attn_kernel<<<12288 / 4, 128, 0, stream>>>(qbuf, kbuf, vTbuf, relb, ywbuf /*attn_out*/);

  // 6. proj + window-reverse + residual -> y (=d_out)   (6272 x 3 waves)
  proj_gemm_kernel<<<18816 / 8, 256, 0, stream>>>(ywbuf, projT, proj_b, x, y);

  // 7. fused LN2 + MLP + residual, in-place on d_out
  mlp_kernel<<<TOK / 16, 256, 0, stream>>>(y, n2w, n2b, fc1T, fc1_b, fc2T, fc2_b);
}
